// KalmanPredictorMambaLatentAdjustment_14164802142683
// MI455X (gfx1250) — matrix operand order repaired and checked by reading, not compiled, not run on hardware
//
#include <hip/hip_runtime.h>

typedef __attribute__((ext_vector_type(16))) _Float16 v16h;
typedef __attribute__((ext_vector_type(8)))  float    v8f;

union Frag16 { v16h v; _Float16 e[16]; };
union Frag8f { v8f  v; float     e[8];  };

#define BB    2
#define DIMC  48
#define HIDC  36
#define DI    72
#define DST   16
#define DTRNK 3
#define KD    4
#define HH    64
#define WW    64
#define LL    4096
#define XPC   (DTRNK + 2 * DST)   // 35

// ---------------------------------------------------------------------------
// Fully specialized WMMA GEMM: C = A * B (+bias), f16 in / f32 acc.
// 128 threads = 4 waves; block computes a 16x64 C tile (one 16x16 per wave).
// A tile (16x32) staged once per block; B staged transposed (BsT[n][k]) so
// fragment gathers are contiguous -> ds_load_b128. Staging uses float4
// (global_load_b128) loads: only the K direction needs zero-fill, M/N are
// handled purely by clamping the load address. Writeback: when the M dim of
// C is contiguous, a lane's 8 outputs are contiguous -> merged b128 stores
// under one hoisted N check; when the N dim is contiguous, the tile bounces
// through LDS for a coalesced float4 row writeback (2 stores/thread).
// ---------------------------------------------------------------------------
template<int M_, int N_, int K_,
         long A_RS, long A_CS, long A_BS, int A_BM,
         long B_RS, long B_CS, long B_BS, int B_BM,
         long C_RS, long C_CS, long C_BS, int BIAS_>
__global__ __launch_bounds__(128) void gemm_wmma_t(
    const float* __restrict__ A, const float* __restrict__ Bw,
    float* __restrict__ C, const float* __restrict__ bias)
{
  __shared__ __align__(16) char smem[16 * 40 * 2 + 64 * 40 * 2];  // 6400 B
  _Float16 (&As)[16][40]  = *reinterpret_cast<_Float16(*)[16][40]>(smem);
  _Float16 (&BsT)[64][40] = *reinterpret_cast<_Float16(*)[64][40]>(smem + 1280);
  const int bz = blockIdx.z;
  A  += (long)(bz % A_BM) * A_BS;
  Bw += (long)(bz % B_BM) * B_BS;
  C  += (long)bz * C_BS;
  const int tm  = blockIdx.x * 16;
  const int tn0 = blockIdx.y * 64;
  const int t    = threadIdx.x;
  const int wave = t >> 5;
  const int lane = t & 31;
  const int half = lane >> 4, lrow = lane & 15;
  const int tn = tn0 + wave * 16;

  v8f acc = {};
  constexpr int KCH = (K_ + 31) / 32;
  #pragma unroll
  for (int kc = 0; kc < KCH; ++kc) {
    const int k0 = kc * 32;
    const bool fullK = (k0 + 32 <= K_);      // compile-time per unrolled chunk
    if (kc + 1 < KCH)   // prefetch next A K-chunk -> global_prefetch_b8
      __builtin_prefetch(&A[(long)tm * A_RS + (long)(k0 + 32) * A_CS], 0, 1);

    // ---- stage A tile 16x32: one float4 (b128) load per thread ----
    {
      const int idx = t * 4;
      if constexpr (A_CS == 1) {             // K contiguous in memory
        const int r  = idx >> 5;
        const int kk = idx & 31;
        const int gm  = min(tm + r, M_ - 1);
        const int gk0 = k0 + kk;
        const int gkc = min(gk0, (K_ - 1) & ~3);
        const float4 v = *(const float4*)&A[(long)gm * A_RS + gkc];
        const float vv[4] = {v.x, v.y, v.z, v.w};
        #pragma unroll
        for (int e = 0; e < 4; ++e)
          As[r][kk + e] = (_Float16)((fullK || (gk0 + e) < K_) ? vv[e] : 0.0f);
      } else {                               // M contiguous in memory
        const int r0 = idx & 15;
        const int kk = idx >> 4;
        const int gk  = min(k0 + kk, K_ - 1);
        const int gmc = min(tm + r0, (M_ - 1) & ~3);
        const float4 v = *(const float4*)&A[(long)gk * A_CS + gmc];
        const float vv[4] = {v.x, v.y, v.z, v.w};
        const _Float16 zf = (_Float16)0.0f;
        const bool okk = fullK || (k0 + kk) < K_;
        #pragma unroll
        for (int e = 0; e < 4; ++e)
          As[r0 + e][kk] = okk ? (_Float16)vv[e] : zf;
      }
    }
    // ---- stage B tile 32x64 transposed: four float4 loads per thread ----
    #pragma unroll
    for (int g = 0; g < 4; ++g) {
      const int idx = t * 16 + g * 4;
      if constexpr (B_CS == 1) {             // N contiguous in memory
        const int nn0 = idx & 63;
        const int kk  = idx >> 6;
        const int gk  = min(k0 + kk, K_ - 1);
        const int gnc = min(tn0 + nn0, (N_ - 1) & ~3);
        const float4 v = *(const float4*)&Bw[(long)gk * B_RS + gnc];
        const float vv[4] = {v.x, v.y, v.z, v.w};
        const bool okk = fullK || (k0 + kk) < K_;
        #pragma unroll
        for (int e = 0; e < 4; ++e)
          BsT[nn0 + e][kk] = okk ? (_Float16)vv[e] : (_Float16)0.0f;
      } else {                               // K contiguous in memory
        const int kk0 = idx & 31;
        const int nn  = idx >> 5;
        const int gn  = min(tn0 + nn, N_ - 1);
        const int gk0 = k0 + kk0;
        const int gkc = min(gk0, (K_ - 1) & ~3);
        const float4 v = *(const float4*)&Bw[(long)gn * B_CS + gkc];
        const float vv[4] = {v.x, v.y, v.z, v.w};
        #pragma unroll
        for (int e = 0; e < 4; ++e)
          BsT[nn][kk0 + e] =
              (_Float16)((fullK || (gk0 + e) < K_) ? vv[e] : 0.0f);
      }
    }
    __syncthreads();
    Frag16 af, bf;
    // A frag (ISA 7.12.2): lanes 0-15 Kbase 0, lanes 16-31 Kbase 8;
    // both 8-element runs contiguous in As -> merged ds_load_b128.
    #pragma unroll
    for (int j = 0; j < 16; ++j) {
      int kk = (j < 8) ? (half * 8 + j) : (16 + half * 8 + (j - 8));
      af.e[j] = As[lrow][kk];
    }
    // B frag: same K order as A (Kbase half*8, then 16 + half*8); contiguous runs in BsT.
    #pragma unroll
    for (int j = 0; j < 16; ++j)
      bf.e[j] = BsT[wave * 16 + lrow][(j < 8) ? (half * 8 + j) : (16 + half * 8 + (j - 8))];
    acc = __builtin_amdgcn_wmma_f32_16x16x32_f16(false, af.v, false, bf.v,
                                                 (short)0, acc, false, false);
    __syncthreads();   // also protects the Cs writeback reuse of smem below
  }

  Frag8f o; o.v = acc;
  constexpr bool MFULL = (M_ % 16 == 0);   // grid covers M exactly when true
  if constexpr (C_CS == 1 && (N_ % 64 == 0)) {
    // ---- LDS-bounce writeback: coalesced float4 row stores ----
    float* Cs = reinterpret_cast<float*>(smem);       // [16][68] f32
    #pragma unroll
    for (int r = 0; r < 8; ++r) {
      float v = o.e[r];
      if constexpr (BIAS_) v += bias[tn + lrow];
      Cs[(half * 8 + r) * 68 + wave * 16 + lrow] = v;
    }
    __syncthreads();
    #pragma unroll
    for (int g = 0; g < 2; ++g) {
      const int gi  = t * 2 + g;       // 256 float4 slots in 16x64 tile
      const int row = gi >> 4;
      const int c4  = gi & 15;
      const int gm  = tm + row;
      if (MFULL || gm < M_) {
        const float4 v = *(const float4*)&Cs[row * 68 + c4 * 4];
        *(float4*)&C[(long)gm * C_RS + (long)(tn0 + c4 * 4)] = v;
      }
    }
  } else {
    // ---- direct writeback: lane's 8 outputs contiguous when C_RS==1 ----
    const int gn = tn + lrow;
    float bv = 0.f;
    if constexpr (BIAS_) bv = bias[min(gn, N_ - 1)];
    if (gn < N_) {
      #pragma unroll
      for (int r = 0; r < 8; ++r) {
        const int gm = tm + half * 8 + r;
        if (MFULL || gm < M_) {
          float v = o.e[r];
          if constexpr (BIAS_) v += bv;
          C[(long)gm * C_RS + (long)gn * C_CS] = v;
        }
      }
    }
  }
}

// ---------------------------------------------------------------------------
// Pre-stage 1: 1x1 conv (48->36) + bias + leaky_relu(0.01) + channel LayerNorm
// ---------------------------------------------------------------------------
__global__ void pre1_conv_lrelu_ln(const float* __restrict__ x,
                                   const float* __restrict__ w1,
                                   const float* __restrict__ b1,
                                   const float* __restrict__ g,
                                   const float* __restrict__ be,
                                   float* __restrict__ z1)
{
  int idx = blockIdx.x * blockDim.x + threadIdx.x;
  if (idx >= BB * LL) return;
  int b = idx / LL, l = idx % LL;
  const float* xp = x + (long)b * DIMC * LL + l;
  float acc[HIDC];
  #pragma unroll
  for (int o = 0; o < HIDC; ++o) acc[o] = b1[o];
  for (int c = 0; c < DIMC; ++c) {
    float xv = xp[(long)c * LL];
    #pragma unroll
    for (int o = 0; o < HIDC; ++o) acc[o] += w1[o * DIMC + c] * xv;
  }
  float mu = 0.f;
  #pragma unroll
  for (int o = 0; o < HIDC; ++o) {
    float v = acc[o];
    v = v > 0.f ? v : 0.01f * v;
    acc[o] = v; mu += v;
  }
  mu *= (1.0f / HIDC);
  float var = 0.f;
  #pragma unroll
  for (int o = 0; o < HIDC; ++o) { float d = acc[o] - mu; var += d * d; }
  var *= (1.0f / HIDC);
  float rs = rsqrtf(var + 1e-5f);
  float* zp = z1 + (long)b * HIDC * LL + l;
  #pragma unroll
  for (int o = 0; o < HIDC; ++o)
    zp[(long)o * LL] = (acc[o] - mu) * rs * g[o] + be[o];
}

// 3x3 conv (36->36, SAME) + sigmoid. One thread per (b,o,l).
__global__ void conv3_sigmoid(const float* __restrict__ z1,
                              const float* __restrict__ w3,
                              const float* __restrict__ b3,
                              float* __restrict__ z)
{
  int idx = blockIdx.x * blockDim.x + threadIdx.x;
  if (idx >= BB * HIDC * LL) return;
  int l = idx % LL; int t = idx / LL;
  int o = t % HIDC; int b = t / HIDC;
  int h = l / WW, w = l % WW;
  float s = b3[o];
  const float* zb = z1 + (long)b * HIDC * LL;
  for (int ci = 0; ci < HIDC; ++ci) {
    const float* zc = zb + (long)ci * LL;
    const float* wp = w3 + (long)(o * HIDC + ci) * 9;
    #pragma unroll
    for (int kh = 0; kh < 3; ++kh) {
      int hh = h + kh - 1; if (hh < 0 || hh >= HH) continue;
      #pragma unroll
      for (int kw = 0; kw < 3; ++kw) {
        int w2 = w + kw - 1; if (w2 < 0 || w2 >= WW) continue;
        s += zc[hh * WW + w2] * wp[kh * 3 + kw];
      }
    }
  }
  z[idx] = 1.0f / (1.0f + __expf(-s));
}

// Depthwise 3x3 conv on xin (= xz channels 0..71) + bias + SiLU.
__global__ void dwconv_silu(const float* __restrict__ xz,
                            const float* __restrict__ cw,
                            const float* __restrict__ cb,
                            float* __restrict__ xconv)
{
  int idx = blockIdx.x * blockDim.x + threadIdx.x;
  if (idx >= BB * DI * LL) return;
  int l = idx % LL; int t = idx / LL;
  int d = t % DI; int b = t / DI;
  int h = l / WW, w = l % WW;
  const float* xp = xz + ((long)b * (2 * DI) + d) * LL;
  float s = cb[d];
  #pragma unroll
  for (int kh = 0; kh < 3; ++kh) {
    int hh = h + kh - 1; if (hh < 0 || hh >= HH) continue;
    #pragma unroll
    for (int kw = 0; kw < 3; ++kw) {
      int w2 = w + kw - 1; if (w2 < 0 || w2 >= WW) continue;
      s += xp[hh * WW + w2] * cw[d * 9 + kh * 3 + kw];
    }
  }
  xconv[idx] = s * (1.0f / (1.0f + __expf(-s)));
}

// Scatter xconv into the 4 scan directions: row, col-major, and their flips.
__global__ void build_xs(const float* __restrict__ xconv, float* __restrict__ xs)
{
  int idx = blockIdx.x * blockDim.x + threadIdx.x;
  if (idx >= BB * DI * LL) return;
  int l = idx % LL; int t = idx / LL;
  int d = t % DI; int b = t / DI;
  float v = xconv[idx];
  int h = l / WW, w = l % WW;
  int lt = w * HH + h;
  long s0 = ((long)(b * KD + 0) * DI + d) * LL;
  long s1 = ((long)(b * KD + 1) * DI + d) * LL;
  long s2 = ((long)(b * KD + 2) * DI + d) * LL;
  long s3 = ((long)(b * KD + 3) * DI + d) * LL;
  xs[s0 + l] = v;
  xs[s1 + lt] = v;
  xs[s2 + (LL - 1 - l)] = v;
  xs[s3 + (LL - 1 - lt)] = v;
}

// delta = softplus(dts @ dt_w + dt_b); dts = x_dbl rows 0..2.
__global__ void delta_softplus(const float* __restrict__ xdbl,
                               const float* __restrict__ dtw,
                               const float* __restrict__ dtb,
                               float* __restrict__ delta)
{
  int idx = blockIdx.x * blockDim.x + threadIdx.x;
  if (idx >= BB * KD * DI * LL) return;
  int l = idx % LL; int t = idx / LL;
  int d = t % DI; int bk = t / DI; int k = bk % KD;
  const float* xp = xdbl + (long)bk * XPC * LL + l;
  int kd = k * DI + d;
  float s = dtb[kd];
  #pragma unroll
  for (int r = 0; r < DTRNK; ++r) s += xp[(long)r * LL] * dtw[kd * DTRNK + r];
  delta[idx] = (s > 20.f) ? s : log1pf(__expf(s));
}

// Sequential selective scan. Block = one (b,k); lanes 0..71 own a d-channel
// with h[16]/A[16] register-resident; lanes 96..127 stage B_l/C_l into a
// ping-pong LDS buffer (one barrier per timestep).
__global__ void selective_scan(const float* __restrict__ delta,
                               const float* __restrict__ xs,
                               const float* __restrict__ xdbl,
                               const float* __restrict__ Alogs,
                               const float* __restrict__ Dvec,
                               float* __restrict__ ys)
{
  __shared__ float sB[2][DST];
  __shared__ float sC[2][DST];
  const int bk = blockIdx.x;
  const int k  = bk % KD;
  const int tid = threadIdx.x;
  float hst[DST]; float An[DST];
  float Dval = 0.f;
  const float* dp = nullptr; const float* up = nullptr; float* yp = nullptr;
  if (tid < DI) {
    int kd = k * DI + tid;
    #pragma unroll
    for (int n = 0; n < DST; ++n) {
      An[n]  = -__expf(Alogs[kd * DST + n]);
      hst[n] = 0.f;
    }
    Dval = Dvec[kd];
    long off = ((long)bk * DI + tid) * LL;
    dp = delta + off; up = xs + off; yp = ys + off;
  }
  const float* xb = xdbl + (long)bk * XPC * LL;
  if (tid >= 96) {            // prime buffer 0 with l=0
    int j = tid - 96;
    if (j < DST) sB[0][j] = xb[(long)(DTRNK + j) * LL];
    else         sC[0][j - DST] = xb[(long)(DTRNK + DST + (j - DST)) * LL];
  }
  __syncthreads();
  for (int l = 0; l < LL; ++l) {
    const int cur = l & 1;
    if (tid >= 96 && (l + 1) < LL) {   // stage l+1 into the other buffer
      int j = tid - 96;
      if (j < DST) sB[cur ^ 1][j] = xb[(long)(DTRNK + j) * LL + (l + 1)];
      else sC[cur ^ 1][j - DST] = xb[(long)(DTRNK + DST + (j - DST)) * LL + (l + 1)];
    }
    if (tid < DI) {
      float dl = dp[l];
      float ul = up[l];
      float du = dl * ul;
      float y = 0.f;
      #pragma unroll
      for (int n = 0; n < DST; ++n) {
        float hn = __expf(dl * An[n]) * hst[n] + du * sB[cur][n];
        hst[n] = hn;
        y += hn * sC[cur][n];
      }
      yp[l] = y + Dval * ul;
    }
    __syncthreads();
  }
}

// Merge the 4 scanned directions (with flips / HW transposes).
__global__ void combine_dirs(const float* __restrict__ ys, float* __restrict__ yc)
{
  int idx = blockIdx.x * blockDim.x + threadIdx.x;
  if (idx >= BB * DI * LL) return;
  int l = idx % LL; int t = idx / LL;
  int d = t % DI; int b = t / DI;
  int h = l / WW, w = l % WW;
  int lt = w * HH + h;
  long b0 = ((long)(b * KD + 0) * DI + d) * LL;
  long b1 = ((long)(b * KD + 1) * DI + d) * LL;
  long b2 = ((long)(b * KD + 2) * DI + d) * LL;
  long b3 = ((long)(b * KD + 3) * DI + d) * LL;
  yc[idx] = ys[b0 + l] + ys[b2 + (LL - 1 - l)]
          + ys[b3 + (LL - 1 - lt)] + ys[b1 + lt];
}

// LayerNorm over D_INNER + SiLU(z-gate) multiply; emits (B,L,72) for out_proj.
__global__ void ln_gate(const float* __restrict__ yc,
                        const float* __restrict__ xz,
                        const float* __restrict__ g,
                        const float* __restrict__ be,
                        float* __restrict__ yg)
{
  int idx = blockIdx.x * blockDim.x + threadIdx.x;
  if (idx >= BB * LL) return;
  int b = idx / LL, l = idx % LL;
  const float* yp = yc + (long)b * DI * LL + l;
  float mu = 0.f;
  for (int d = 0; d < DI; ++d) mu += yp[(long)d * LL];
  mu *= (1.0f / DI);
  float var = 0.f;
  for (int d = 0; d < DI; ++d) { float v = yp[(long)d * LL] - mu; var += v * v; }
  var *= (1.0f / DI);
  float rs = rsqrtf(var + 1e-5f);
  const float* zp = xz + ((long)b * (2 * DI) + DI) * LL + l;
  float* op = yg + ((long)b * LL + l) * DI;
  for (int d = 0; d < DI; ++d) {
    float v = (yp[(long)d * LL] - mu) * rs * g[d] + be[d];
    float zg = zp[(long)d * LL];
    op[d] = v * (zg / (1.0f + __expf(-zg)));
  }
}

// z / exp(-sigma) + bias  ==  z * exp(sigma) + bias
__global__ void gate_combine(const float* __restrict__ z,
                             const float* __restrict__ sig,
                             const float* __restrict__ bia,
                             float* __restrict__ zc)
{
  int idx = blockIdx.x * blockDim.x + threadIdx.x;
  if (idx >= BB * HIDC * LL) return;
  zc[idx] = z[idx] * __expf(sig[idx]) + bia[idx];
}

// ---------------------------------------------------------------------------
// Host-side orchestration
// ---------------------------------------------------------------------------
struct SS2DParams {
  const float *in_proj_w, *conv_w, *conv_b, *x_proj_w, *dt_w, *dt_b,
              *A_logs, *Ds, *ln_g, *ln_b, *out_proj_w;
};

static void run_ss2d(hipStream_t s, const float* z, const SS2DParams& p,
                     float* xz, float* xconv, float* xs, float* xdbl,
                     float* delta, float* ys, float* yc, float* yg,
                     float* out)
{
  // in_proj: per-b GEMM [4096 x 36] * [36 x 144]; A = z channel-major.
  gemm_wmma_t<LL, 2 * DI, HIDC,
              1L, (long)LL, (long)HIDC * LL, BB,
              1L, (long)HIDC, 0L, 1,
              1L, (long)LL, (long)(2 * DI) * LL, 0>
      <<<dim3(LL / 16, (2 * DI + 63) / 64, BB), 128, 0, s>>>(
          z, p.in_proj_w, xz, nullptr);
  int n1 = BB * DI * LL;
  dwconv_silu<<<(n1 + 255) / 256, 256, 0, s>>>(xz, p.conv_w, p.conv_b, xconv);
  build_xs<<<(n1 + 255) / 256, 256, 0, s>>>(xconv, xs);
  // x_dbl: per (b,k) GEMM [35 x 72] * [72 x 4096]; weight offset keyed on k.
  gemm_wmma_t<XPC, LL, DI,
              (long)DI, 1L, (long)XPC * DI, KD,
              (long)LL, 1L, (long)DI * LL, BB * KD,
              (long)LL, 1L, (long)XPC * LL, 0>
      <<<dim3((XPC + 15) / 16, LL / 64, BB * KD), 128, 0, s>>>(
          p.x_proj_w, xs, xdbl, nullptr);
  int n2 = BB * KD * DI * LL;
  delta_softplus<<<(n2 + 255) / 256, 256, 0, s>>>(xdbl, p.dt_w, p.dt_b, delta);
  selective_scan<<<BB * KD, 128, 0, s>>>(delta, xs, xdbl, p.A_logs, p.Ds, ys);
  combine_dirs<<<(n1 + 255) / 256, 256, 0, s>>>(ys, yc);
  int n3 = BB * LL;
  ln_gate<<<(n3 + 255) / 256, 256, 0, s>>>(yc, xz, p.ln_g, p.ln_b, yg);
  // out_proj: per-b GEMM [4096 x 72] * [72 x 36] -> (B,36,L)
  gemm_wmma_t<LL, HIDC, DI,
              (long)DI, 1L, (long)LL * DI, BB,
              1L, (long)DI, 0L, 1,
              1L, (long)LL, (long)HIDC * LL, 0>
      <<<dim3(LL / 16, (HIDC + 63) / 64, BB), 128, 0, s>>>(
          yg, p.out_proj_w, out, nullptr);
}

extern "C" void kernel_launch(void* const* d_in, const int* in_sizes, int n_in,
                              void* d_out, int out_size, void* d_ws, size_t ws_size,
                              hipStream_t stream)
{
  (void)in_sizes; (void)n_in; (void)out_size; (void)ws_size;
  const float* x   = (const float*)d_in[0];
  const float* w1  = (const float*)d_in[1];
  const float* b1  = (const float*)d_in[2];
  const float* plg = (const float*)d_in[3];
  const float* plb = (const float*)d_in[4];
  const float* w3  = (const float*)d_in[5];
  const float* b3  = (const float*)d_in[6];
  const float* wo  = (const float*)d_in[7];
  const float* bo  = (const float*)d_in[8];
  SS2DParams ms = { (const float*)d_in[9],  (const float*)d_in[10],
                    (const float*)d_in[11], (const float*)d_in[12],
                    (const float*)d_in[13], (const float*)d_in[14],
                    (const float*)d_in[15], (const float*)d_in[16],
                    (const float*)d_in[17], (const float*)d_in[18],
                    (const float*)d_in[19] };
  SS2DParams mb = { (const float*)d_in[20], (const float*)d_in[21],
                    (const float*)d_in[22], (const float*)d_in[23],
                    (const float*)d_in[24], (const float*)d_in[25],
                    (const float*)d_in[26], (const float*)d_in[27],
                    (const float*)d_in[28], (const float*)d_in[29],
                    (const float*)d_in[30] };

  float* ws = (float*)d_ws;
  size_t off = 0;
  auto carve = [&](size_t n) { float* p = ws + off; off += n; return p; };
  float* z1    = carve((size_t)BB * HIDC * LL);
  float* z     = carve((size_t)BB * HIDC * LL);
  float* xz    = carve((size_t)BB * 2 * DI * LL);
  float* xconv = carve((size_t)BB * DI * LL);
  float* xs    = carve((size_t)BB * KD * DI * LL);
  float* xdbl  = carve((size_t)BB * KD * XPC * LL);
  float* delta = carve((size_t)BB * KD * DI * LL);
  float* ys    = carve((size_t)BB * KD * DI * LL);
  float* yc    = carve((size_t)BB * DI * LL);
  float* yg    = carve((size_t)BB * LL * DI);
  float* sigma = carve((size_t)BB * HIDC * LL);
  float* biasv = carve((size_t)BB * HIDC * LL);
  float* zc    = z1;   // z1 dead after conv3_sigmoid -> reuse

  int np = BB * LL;
  pre1_conv_lrelu_ln<<<(np + 255) / 256, 256, 0, stream>>>(x, w1, b1, plg, plb, z1);
  int nz = BB * HIDC * LL;
  conv3_sigmoid<<<(nz + 255) / 256, 256, 0, stream>>>(z1, w3, b3, z);

  run_ss2d(stream, z, ms, xz, xconv, xs, xdbl, delta, ys, yc, yg, sigma);
  run_ss2d(stream, z, mb, xz, xconv, xs, xdbl, delta, ys, yc, yg, biasv);

  gate_combine<<<(nz + 255) / 256, 256, 0, stream>>>(z, sigma, biasv, zc);

  // Final 1x1 conv (36 -> 48) + bias, straight into d_out (B,48,H,W).
  gemm_wmma_t<LL, DIMC, HIDC,
              1L, (long)LL, (long)HIDC * LL, BB,
              1L, (long)HIDC, 0L, 1,
              1L, (long)LL, (long)DIMC * LL, 1>
      <<<dim3(LL / 16, (DIMC + 63) / 64, BB), 128, 0, stream>>>(
          zc, wo, (float*)d_out, bo);
}
